// SpatialAttention_72481868087584
// MI455X (gfx1250) — compile-verified, matrix operand order checked by reading, not run on hardware
//
#include <hip/hip_runtime.h>
#include <stdint.h>

// ---------------- CDNA5 WMMA types ----------------
typedef __attribute__((ext_vector_type(16))) __bf16 bf16x16;
typedef __attribute__((ext_vector_type(4)))  __bf16 bf16x4;
typedef __attribute__((ext_vector_type(8)))  float  f32x8;
typedef __attribute__((ext_vector_type(4)))  float  f32x4;
typedef __attribute__((ext_vector_type(4)))  int    i32x4;
typedef __attribute__((ext_vector_type(4)))  unsigned int u32x4;

#define N_NODES 8192
#define F_DIM   256
#define H_DIM   64

union Frag { bf16x16 v; bf16x4 q[4]; uint32_t u[8]; };

// Native f32 -> bf16 (RNE) via hardware cvt; backend picks v_cvt_pk_bf16_f32.
__device__ __forceinline__ void pack8_f32(const float* p, Frag& f, int qb) {
  f32x4 a = *(const f32x4*)(p);
  f32x4 b = *(const f32x4*)(p + 4);
  f.q[qb]     = __builtin_convertvector(a, bf16x4);
  f.q[qb + 1] = __builtin_convertvector(b, bf16x4);
}
// adj entries are exactly 0/1: bf16(1.0) == 0x3F80 -> integer select pack.
__device__ __forceinline__ uint32_t pk_adj(int a, int b) {
  return ((a > 0) ? 0x3F80u : 0u) | ((b > 0) ? 0x3F800000u : 0u);
}
__device__ __forceinline__ void pack4_adj(i32x4 q, uint32_t* u, int& d) {
  u[0] = pk_adj(q[0], q[1]);
  u[1] = pk_adj(q[2], q[3]);
  d += (q[0] > 0) + (q[1] > 0) + (q[2] > 0) + (q[3] > 0);
}
__device__ __forceinline__ f32x8 wmma_bf16(const Frag& a, const Frag& b, f32x8 c) {
  return __builtin_amdgcn_wmma_f32_16x16x32_bf16(false, a.v, false, b.v,
                                                 (short)0, c, false, false);
}

// ---------------- Kernel 1: h = x @ Wt^T + bt ; also hT (bf16, transposed) ----
__global__ __launch_bounds__(128) void k_transform(
    const float* __restrict__ x, const float* __restrict__ Wt,
    const float* __restrict__ bt, float* __restrict__ h,
    __bf16* __restrict__ hT)
{
  const int lane  = threadIdx.x & 31;
  const int wave  = threadIdx.x >> 5;
  const int lrow  = lane & 15;
  const int half8 = (lane >> 4) << 3;            // 0 or 8 (k/M half select)
  const int r0    = (blockIdx.x * 4 + wave) * 16;
  const int row   = r0 + lrow;

  f32x8 acc[4] = {};
  for (int kb = 0; kb < F_DIM; kb += 32) {
    const int k0 = kb + half8;
    Frag a;
    {
      const float* p = x + row * F_DIM + k0;
      pack8_f32(p,      a, 0);                   // K = k0..k0+7
      pack8_f32(p + 16, a, 2);                   // K = k0+16..k0+23
    }
    #pragma unroll
    for (int nt = 0; nt < 4; ++nt) {
      const int col = nt * 16 + lrow;            // B[k,n] = Wt[n,k] (contig in k)
      const float* p = Wt + col * F_DIM + k0;
      Frag b;
      pack8_f32(p,      b, 0);
      pack8_f32(p + 16, b, 2);
      acc[nt] = wmma_bf16(a, b, acc[nt]);
    }
  }
  #pragma unroll
  for (int nt = 0; nt < 4; ++nt) {
    const int c = nt * 16 + lrow;
    const float bias = bt[c];
    #pragma unroll
    for (int v = 0; v < 8; ++v) {
      const int rg = r0 + v + half8;             // D: M = v + 8*(lane/16)
      const float val = acc[nt][v] + bias;
      h[rg * H_DIM + c] = val;
      hT[c * N_NODES + rg] = (__bf16)val;
    }
  }
}

// ---------------- Kernel 2: column mean of h (deg==0 fallback) ---------------
__global__ __launch_bounds__(256) void k_hmean(const float* __restrict__ h,
                                               float* __restrict__ hmean)
{
  __shared__ float part[4][H_DIM];
  const int c = threadIdx.x & 63;
  const int s = threadIdx.x >> 6;                // 4 row-chunks of 2048
  float a0 = 0.f, a1 = 0.f, a2 = 0.f, a3 = 0.f;
  const int beg = s * 2048;
  for (int i = beg; i < beg + 2048; i += 4) {
    a0 += h[(i + 0) * H_DIM + c];
    a1 += h[(i + 1) * H_DIM + c];
    a2 += h[(i + 2) * H_DIM + c];
    a3 += h[(i + 3) * H_DIM + c];
  }
  part[s][c] = (a0 + a1) + (a2 + a3);
  __syncthreads();
  if (threadIdx.x < 64)
    hmean[c] = (part[0][c] + part[1][c] + part[2][c] + part[3][c]) *
               (1.0f / (float)N_NODES);
}

// ---------------- Kernel 3: partial adj @ h  (+ row degrees), K-split x2 -----
// Block layout: all 4 waves share the SAME k-half and k-blocks (different row
// tiles) so their hT B-fragment loads are identical -> WGP-cache hits.
__global__ __launch_bounds__(128) void k_aggregate(
    const int* __restrict__ adj, const __bf16* __restrict__ hT,
    float* __restrict__ nbp, int* __restrict__ degp)
{
  const int lane  = threadIdx.x & 31;
  const int wave  = threadIdx.x >> 5;
  const int lrow  = lane & 15;
  const int half8 = (lane >> 4) << 3;
  const int kh    = blockIdx.x >> 7;             // K half (same for whole block)
  const int t     = ((blockIdx.x & 127) << 2) | wave;   // row tile 0..511
  const int r0    = t * 16;
  const int row   = r0 + lrow;
  const int kbeg  = kh * (N_NODES / 2);
  const int kend  = kbeg + (N_NODES / 2);

  f32x8 acc[4] = {};
  int dsum = 0;
  #pragma unroll 2
  for (int kb = kbeg; kb < kend; kb += 32) {
    const int k0 = kb + half8;
    Frag a;
    {
      const int* p = adj + row * N_NODES + k0;   // streamed once: non-temporal
      i32x4 q0 = __builtin_nontemporal_load((const i32x4*)(p));
      i32x4 q1 = __builtin_nontemporal_load((const i32x4*)(p + 4));
      i32x4 q2 = __builtin_nontemporal_load((const i32x4*)(p + 16));
      i32x4 q3 = __builtin_nontemporal_load((const i32x4*)(p + 20));
      pack4_adj(q0, &a.u[0], dsum);
      pack4_adj(q1, &a.u[2], dsum);
      pack4_adj(q2, &a.u[4], dsum);
      pack4_adj(q3, &a.u[6], dsum);
    }
    #pragma unroll
    for (int nt = 0; nt < 4; ++nt) {
      const int col = nt * 16 + lrow;            // B[k,c] = hT[c][k] (contig in k)
      const __bf16* p = hT + col * N_NODES + k0;
      Frag b;
      *(u32x4*)(&b.u[0]) = *(const u32x4*)(p);        // K = k0..k0+7
      *(u32x4*)(&b.u[4]) = *(const u32x4*)(p + 16);   // K = k0+16..k0+23
      acc[nt] = wmma_bf16(a, b, acc[nt]);
    }
  }
  // lanes L and L^16 cover complementary k offsets of the same row
  dsum += __shfl_xor(dsum, 16);
  if (lane < 16) degp[kh * N_NODES + row] = dsum;

  float* out = nbp + (size_t)kh * (N_NODES * H_DIM);
  #pragma unroll
  for (int nt = 0; nt < 4; ++nt) {
    const int c = nt * 16 + lrow;
    #pragma unroll
    for (int v = 0; v < 8; ++v)
      out[(r0 + v + half8) * H_DIM + c] = acc[nt][v];
  }
}

// ---------------- Kernel 4: neighbor = (p0+p1)/deg  (or hmean if deg==0) -----
__global__ __launch_bounds__(256) void k_combine(
    const float* __restrict__ nbp, const int* __restrict__ degp,
    const float* __restrict__ hmean, float* __restrict__ nb)
{
  const int i   = blockIdx.x * 256 + threadIdx.x;   // 0..N*H-1
  const int row = i >> 6;
  const int c   = i & 63;
  const int d   = degp[row] + degp[N_NODES + row];
  const float s = nbp[i] + nbp[(size_t)N_NODES * H_DIM + i];
  nb[i] = (d > 0) ? s / (float)d : hmean[c];
}

// ---------------- Kernel 5: tf = nb @ Wp^T + bp ; fused residual + LayerNorm -
__global__ __launch_bounds__(64) void k_project_ln(
    const float* __restrict__ nb, const float* __restrict__ Wp,
    const float* __restrict__ bp, const float* __restrict__ x,
    const float* __restrict__ gamma, const float* __restrict__ beta,
    float* __restrict__ out)
{
  __shared__ float ylds[2][16 * F_DIM];            // 32 KB
  const int lane  = threadIdx.x & 31;
  const int wave  = threadIdx.x >> 5;
  const int lrow  = lane & 15;
  const int half8 = (lane >> 4) << 3;
  const int r0    = (blockIdx.x * 2 + wave) * 16;
  const int row   = r0 + lrow;

  f32x8 acc[16] = {};
  #pragma unroll
  for (int kb = 0; kb < H_DIM; kb += 32) {
    const int k0 = kb + half8;
    Frag a;
    {
      const float* p = nb + row * H_DIM + k0;
      pack8_f32(p,      a, 0);
      pack8_f32(p + 16, a, 2);
    }
    #pragma unroll
    for (int nt = 0; nt < 16; ++nt) {
      const int col = nt * 16 + lrow;              // B[k,n] = Wp[n,k]
      const float* p = Wp + col * H_DIM + k0;
      Frag b;
      pack8_f32(p,      b, 0);
      pack8_f32(p + 16, b, 2);
      acc[nt] = wmma_bf16(a, b, acc[nt]);
    }
  }
  float* my = ylds[wave];
  #pragma unroll
  for (int nt = 0; nt < 16; ++nt) {
    const int c = nt * 16 + lrow;
    const float bias = bp[c];
    #pragma unroll
    for (int v = 0; v < 8; ++v)
      my[(v + half8) * F_DIM + c] = acc[nt][v] + bias;
  }
  __syncthreads();

  for (int m = 0; m < 16; ++m) {
    const float* xr = x + (size_t)(r0 + m) * F_DIM;
    float s = 0.f, q = 0.f;
    for (int c = lane; c < F_DIM; c += 32) {       // lane touches only its cols
      const float yv = my[m * F_DIM + c] + xr[c];  // residual
      my[m * F_DIM + c] = yv;
      s += yv; q += yv * yv;
    }
    #pragma unroll
    for (int o = 16; o > 0; o >>= 1) { s += __shfl_xor(s, o); q += __shfl_xor(q, o); }
    const float mu  = s * (1.0f / F_DIM);
    const float var = q * (1.0f / F_DIM) - mu * mu;
    const float inv = rsqrtf(var + 1e-5f);
    float* orow = out + (size_t)(r0 + m) * F_DIM;
    for (int c = lane; c < F_DIM; c += 32)
      orow[c] = gamma[c] * (my[m * F_DIM + c] - mu) * inv + beta[c];
  }
}

// ---------------- Host launch ------------------------------------------------
extern "C" void kernel_launch(void* const* d_in, const int* in_sizes, int n_in,
                              void* d_out, int out_size, void* d_ws, size_t ws_size,
                              hipStream_t stream)
{
  (void)in_sizes; (void)n_in; (void)out_size; (void)ws_size;
  const float* x     = (const float*)d_in[0];
  const int*   adj   = (const int*)  d_in[1];
  const float* Wt    = (const float*)d_in[2];
  const float* bt    = (const float*)d_in[3];
  // d_in[4] = Wa, d_in[5] = ba: provably cancel in the row-constant softmax.
  const float* Wp    = (const float*)d_in[6];
  const float* bp    = (const float*)d_in[7];
  const float* gamma = (const float*)d_in[8];
  const float* beta  = (const float*)d_in[9];
  float* out = (float*)d_out;

  char* ws = (char*)d_ws;                    // ~9.6 MB total
  float*   h     = (float*)  (ws);                          // 2 MB
  __bf16*  hT    = (__bf16*) (ws + (2u << 20));             // 1 MB
  float*   nbp   = (float*)  (ws + (3u << 20));             // 4 MB (2 partials)
  int*     degp  = (int*)    (ws + (7u << 20));             // 64 KB (2 partials)
  float*   nb    = (float*)  (ws + (7u << 20) + (1u << 16));// 2 MB
  float*   hmean = (float*)  (ws + (9u << 20) + (1u << 16));// 256 B

  k_transform<<<128, 128, 0, stream>>>(x, Wt, bt, h, hT);
  k_hmean    <<<1, 256, 0, stream>>>(h, hmean);
  k_aggregate<<<256, 128, 0, stream>>>(adj, hT, nbp, degp);
  k_combine  <<<(N_NODES * H_DIM) / 256, 256, 0, stream>>>(nbp, degp, hmean, nb);
  k_project_ln<<<256, 64, 0, stream>>>(nb, Wp, bp, x, gamma, beta, out);
}
